// SepConvNextBlock_38663295598620
// MI455X (gfx1250) — matrix operand order repaired and checked by reading, not compiled, not run on hardware
//
#include <hip/hip_runtime.h>
#include <math.h>

typedef __attribute__((ext_vector_type(16))) _Float16 v16h;
typedef __attribute__((ext_vector_type(8)))  float    v8f;

#define N_PTS   256   // m / n
#define N_OR    16    // o / p
#define N_CH    64    // c
#define N_BASIS 32
#define HID     256   // WIDEN * C

#define HN_LD 72      // padded f16 row stride (bank-conflict-free)
#define Y1_LD 264     // padded f16 row stride (bank-conflict-free)

// ---------------- WMMA fragment loaders (contiguous k order, same for A and B) ----------------

// A (16x32 f16, M x K): lane holds row M = lane%16; element e:
//   K = (lane<16?0:16) + e
// Here A[M,K] = KB[m, n0+K, o, kbase+M]  (n-stride = 16*32 floats)
__device__ __forceinline__ v16h a_frag_kb(const float* __restrict__ base,
                                          int n0, int krow, int lane) {
  v16h a;
  const int lh = (lane < 16) ? 0 : 16;
#pragma unroll
  for (int e = 0; e < 16; ++e) {
    const int K = lh + e;
    a[e] = (_Float16)__builtin_nontemporal_load(
        &base[(n0 + K) * (N_OR * N_BASIS) + krow]);
  }
  return a;
}

// B (32x16 f16, K x N): lane holds col N = lane%16; K = (lane<16?0:16) + e
// Here B[K,N] = x[n0+K, o, c]  (n-stride = 16*64 floats)
__device__ __forceinline__ v16h b_frag_x(const float* __restrict__ xo,
                                         int n0, int c, int lane) {
  v16h b;
  const int off = (lane < 16) ? 0 : 16;
#pragma unroll
  for (int e = 0; e < 16; ++e)
    b[e] = (_Float16)xo[(n0 + off + e) * (N_OR * N_CH) + c];
  return b;
}

// A fragment from an f16 row-major LDS matrix (row = M, col = K)
__device__ __forceinline__ v16h a_frag_lds(const _Float16* H, int ld, int k0, int lane) {
  v16h a;
  const int row = lane & 15;
  const int lh = (lane < 16) ? 0 : 16;
#pragma unroll
  for (int e = 0; e < 16; ++e) {
    const int K = lh + e;
    a[e] = H[row * ld + k0 + K];
  }
  return a;
}

// B fragment from an f16 row-major global matrix (row = K, col = N)
__device__ __forceinline__ v16h b_frag_h(const _Float16* __restrict__ W, int ld,
                                         int k0, int c, int lane) {
  v16h b;
  const int off = (lane < 16) ? 0 : 16;
#pragma unroll
  for (int e = 0; e < 16; ++e)
    b[e] = W[(k0 + off + e) * ld + c];
  return b;
}

__device__ __forceinline__ float gelu_tanh(float t) {
  const float k = 0.7978845608028654f;  // sqrt(2/pi)
  return 0.5f * t * (1.f + tanhf(k * (t + 0.044715f * t * t * t)));
}

// ---------------- prep: rot_kernel (p,o,c) + f16 weight copies ----------------
__global__ __launch_bounds__(256) void sepconv_prep_kernel(
    const float* __restrict__ fiber,  // (16,16,32)
    const float* __restrict__ Wrot,   // (32,64)
    const float* __restrict__ W1,     // (64,256)
    const float* __restrict__ W2,     // (256,64)
    float* __restrict__ rotk,         // (16,16,64)
    _Float16* __restrict__ W1h, _Float16* __restrict__ W2h) {
  const int i = blockIdx.x * 256 + threadIdx.x;  // 0..16383
  const int po = i >> 6;
  const int c = i & 63;
  float s = 0.f;
#pragma unroll
  for (int k = 0; k < N_BASIS; ++k)
    s += fiber[po * N_BASIS + k] * Wrot[k * N_CH + c];
  rotk[i] = s;
  W1h[i] = (_Float16)W1[i];
  W2h[i] = (_Float16)W2[i];
}

// ---------------- fused main kernel: one block per m ----------------
__global__ __launch_bounds__(512) void sepconv_block_kernel(
    const float* __restrict__ x,         // (256,16,64)
    const float* __restrict__ KB,        // (256,256,16,32)
    const float* __restrict__ Wsp,       // (32,64)
    const float* __restrict__ conv_bias, // (64)
    const float* __restrict__ ln_scale, const float* __restrict__ ln_bias,
    const float* __restrict__ b1,        // (256)
    const float* __restrict__ b2,        // (64)
    const float* __restrict__ rotk,      // ws (16,16,64) f32
    const _Float16* __restrict__ W1h,    // ws (64,256) f16
    const _Float16* __restrict__ W2h,    // ws (256,64) f16
    float* __restrict__ out) {           // (256,16,64)
  __shared__ float    h1s[N_OR * N_CH];
  __shared__ _Float16 Hn[N_OR * HN_LD];
  __shared__ _Float16 Y1[N_OR * Y1_LD];

  const int m = blockIdx.x;
  const int w = threadIdx.x >> 5;    // wave id, 0..15
  const int lane = threadIdx.x & 31;
  const int l15 = lane & 15;

  // ======== Stage 1: Y = KB[m,:,o,:]^T @ X_o  then h1[o,c] = sum_k Y*Wsp ========
  {
    const int o = w;
    const float* KBmo = KB + (size_t)m * (N_PTS * N_OR * N_BASIS) + o * N_BASIS;
    const float* xo = x + o * N_CH;
    v8f acc[8] = {};  // [mt*4 + nt], M = k(32) tiles x N = c(64) tiles
    for (int ch = 0; ch < 8; ++ch) {       // K = n, 8 chunks of 32
      const int n0 = ch * 32;
      v16h a0 = a_frag_kb(KBmo, n0, l15, lane);        // k rows 0..15
      v16h a1 = a_frag_kb(KBmo, n0, 16 + l15, lane);   // k rows 16..31
#pragma unroll
      for (int nt = 0; nt < 4; ++nt) {
        v16h b = b_frag_x(xo, n0, nt * 16 + l15, lane);
        acc[nt] = __builtin_amdgcn_wmma_f32_16x16x32_f16(
            false, a0, false, b, (short)0, acc[nt], false, false);
        acc[4 + nt] = __builtin_amdgcn_wmma_f32_16x16x32_f16(
            false, a1, false, b, (short)0, acc[4 + nt], false, false);
      }
    }
    // weighted k-reduction: h1[o,c] = sum_k Y[k,c] * Wsp[k,c]
    const int kh = (lane < 16) ? 0 : 8;  // C/D layout: M = v + (lane<16?0:8)
#pragma unroll
    for (int nt = 0; nt < 4; ++nt) {
      const int c = nt * 16 + l15;
      float p = 0.f;
#pragma unroll
      for (int mt = 0; mt < 2; ++mt)
#pragma unroll
        for (int v = 0; v < 8; ++v)
          p += acc[mt * 4 + nt][v] * Wsp[(mt * 16 + kh + v) * N_CH + c];
      p += __shfl_xor(p, 16, 32);  // fold lanes 16-31 into 0-15 (same c)
      if (lane < 16) h1s[o * N_CH + c] = p;
    }
  }
  __syncthreads();

  // ======== Stage 2a: fiber conv + bias + LayerNorm (wave w owns row p=w) ========
  {
    const int p = w;
    float hv[2];
#pragma unroll
    for (int t = 0; t < 2; ++t) {
      const int c = 2 * lane + t;
      float s = conv_bias[c];
#pragma unroll
      for (int o = 0; o < N_OR; ++o)
        s += h1s[o * N_CH + c] * rotk[(p * N_OR + o) * N_CH + c];
      hv[t] = s;
    }
    float s1 = hv[0] + hv[1];
    float s2 = hv[0] * hv[0] + hv[1] * hv[1];
#pragma unroll
    for (int off = 16; off > 0; off >>= 1) {
      s1 += __shfl_xor(s1, off, 32);
      s2 += __shfl_xor(s2, off, 32);
    }
    const float mu = s1 * (1.f / 64.f);
    const float rstd = rsqrtf(s2 * (1.f / 64.f) - mu * mu + 1e-6f);
#pragma unroll
    for (int t = 0; t < 2; ++t) {
      const int c = 2 * lane + t;
      Hn[p * HN_LD + c] = (_Float16)((hv[t] - mu) * rstd * ln_scale[c] + ln_bias[c]);
    }
  }
  __syncthreads();

  // ======== Stage 2b: MLP layer 1 (16x64 @ 64x256) + GELU, wave w -> j-tile w ========
  {
    v8f acc1 = {};
#pragma unroll
    for (int kc = 0; kc < 2; ++kc) {  // K = c(64), 2 chunks
      v16h a = a_frag_lds(Hn, HN_LD, kc * 32, lane);
      v16h b = b_frag_h(W1h, HID, kc * 32, w * 16 + l15, lane);
      acc1 = __builtin_amdgcn_wmma_f32_16x16x32_f16(
          false, a, false, b, (short)0, acc1, false, false);
    }
    const int j = w * 16 + l15;
    const float bj = b1[j];
    const int ph = (lane < 16) ? 0 : 8;
#pragma unroll
    for (int v = 0; v < 8; ++v)
      Y1[(ph + v) * Y1_LD + j] = (_Float16)gelu_tanh(acc1[v] + bj);
  }
  __syncthreads();

  // ======== Stage 2c: MLP layer 2 (16x256 @ 256x64) + bias + residual ========
  if (w < 4) {  // wave-uniform branch: EXEC stays all-ones inside
    v8f acc2 = {};
    for (int kc = 0; kc < 8; ++kc) {  // K = j(256), 8 chunks
      v16h a = a_frag_lds(Y1, Y1_LD, kc * 32, lane);
      v16h b = b_frag_h(W2h, N_CH, kc * 32, w * 16 + l15, lane);
      acc2 = __builtin_amdgcn_wmma_f32_16x16x32_f16(
          false, a, false, b, (short)0, acc2, false, false);
    }
    const int c = w * 16 + l15;
    const float bc = b2[c];
    const int ph = (lane < 16) ? 0 : 8;
#pragma unroll
    for (int v = 0; v < 8; ++v) {
      const size_t idx = (size_t)m * (N_OR * N_CH) + (ph + v) * N_CH + c;
      out[idx] = acc2[v] + bc + x[idx];
    }
  }
}

extern "C" void kernel_launch(void* const* d_in, const int* in_sizes, int n_in,
                              void* d_out, int out_size, void* d_ws, size_t ws_size,
                              hipStream_t stream) {
  (void)in_sizes; (void)n_in; (void)out_size; (void)ws_size;
  const float* x         = (const float*)d_in[0];
  const float* KB        = (const float*)d_in[1];
  const float* fiber     = (const float*)d_in[2];
  const float* Wsp       = (const float*)d_in[3];
  const float* Wrot      = (const float*)d_in[4];
  const float* conv_bias = (const float*)d_in[5];
  const float* ln_scale  = (const float*)d_in[6];
  const float* ln_bias   = (const float*)d_in[7];
  const float* W1        = (const float*)d_in[8];
  const float* b1        = (const float*)d_in[9];
  const float* W2        = (const float*)d_in[10];
  const float* b2        = (const float*)d_in[11];
  float* out = (float*)d_out;

  // workspace layout: rotk f32 (64KB) | W1h f16 (32KB) | W2h f16 (32KB)
  float*    rotk = (float*)d_ws;
  _Float16* W1h  = (_Float16*)((char*)d_ws + 16384 * sizeof(float));
  _Float16* W2h  = W1h + 64 * 256;

  sepconv_prep_kernel<<<64, 256, 0, stream>>>(fiber, Wrot, W1, W2, rotk, W1h, W2h);
  sepconv_block_kernel<<<N_PTS, 512, 0, stream>>>(
      x, KB, Wsp, conv_bias, ln_scale, ln_bias, b1, b2, rotk, W1h, W2h, out);
}
